// SparseAttention_37383395344553
// MI455X (gfx1250) — compile-verified, matrix operand order checked by reading, not run on hardware
//
#include <hip/hip_runtime.h>
#include <hip/hip_bf16.h>
#include <math.h>

// ---------------------------------------------------------------------------
// Types for CDNA5 WMMA (gfx1250, wave32)
// ---------------------------------------------------------------------------
typedef __attribute__((ext_vector_type(16))) __bf16 v16bf;
typedef __attribute__((ext_vector_type(8)))  __bf16 v8bf;
typedef __attribute__((ext_vector_type(8)))  float  v8f;

#define WMMA_BF16(a, b, c) \
  __builtin_amdgcn_wmma_f32_16x16x32_bf16(false, (a), false, (b), (short)0, (c), false, false)

static __device__ __forceinline__ v8f zero8() {
  v8f z;
#pragma unroll
  for (int i = 0; i < 8; ++i) z[i] = 0.0f;
  return z;
}

// Load a 16x32 bf16 operand fragment (A layout; B uses the same layout on the
// "transposed" storage, i.e. lane&15 selects the N row of W / V^T / K).
// ISA 7.12.2: lane = p*16 + m ; VGPR0..3 hold K = p*8 + {0..7},
// VGPR4..7 hold K = p*8 + 16 + {0..7}  -> two contiguous 16B chunks per lane.
static __device__ __forceinline__ v16bf load_frag(const __bf16* base, int ld, int k0, int lane) {
  const __bf16* p = base + (lane & 15) * ld + k0 + ((lane >> 4) << 3);
  union { v16bf v; v8bf h[2]; } u;
  u.h[0] = *(const v8bf*)(p);
  u.h[1] = *(const v8bf*)(p + 16);
  return u.v;
}

// Async global->LDS copy of one 16-byte chunk (ASYNCcnt-tracked DMA path).
static __device__ __forceinline__ void async_copy_b128(unsigned lds_addr, const void* gptr) {
  unsigned long long g = (unsigned long long)gptr;
  asm volatile("global_load_async_to_lds_b128 %0, %1, off" :: "v"(lds_addr), "v"(g) : "memory");
}

// Monotone key: larger float -> larger unsigned
static __device__ __forceinline__ unsigned fkey(float f) {
  unsigned u = __float_as_uint(f);
  return (u & 0x80000000u) ? ~u : (u | 0x80000000u);
}

// ---------------------------------------------------------------------------
// Problem constants
// ---------------------------------------------------------------------------
#define BATCH   2
#define SEQ     2048
#define DMODEL  768
#define NHEADS  12
#define HDIM    64
#define KKEEP   614            // max(1, int(0.3 * 2048))
#define MROWS   (BATCH * SEQ)  // 4096
#define KT_N    (DMODEL / 32)  // 24 WMMA K-steps for the projections

// ---------------------------------------------------------------------------
// fp32 -> bf16 convert
// ---------------------------------------------------------------------------
__global__ void cvt_bf16_kernel(const float* __restrict__ s, __bf16* __restrict__ d, int n) {
  int i = blockIdx.x * 256 + threadIdx.x;
  if (i < n) d[i] = (__bf16)s[i];
}

// ---------------------------------------------------------------------------
// QKV projection: Y = X @ W^T + b, scatter to head layout.
// One block = (mat, mtile): async-copies the 16x768 A tile to LDS once,
// then each of 8 waves computes a 16(M) x 96(N) strip (6 WMMA accumulators).
// blocks: 3 mats * 256 mtiles = 768
// ---------------------------------------------------------------------------
__global__ void __launch_bounds__(256) qkv_kernel(
    const __bf16* __restrict__ xb,
    const __bf16* __restrict__ Wq, const __bf16* __restrict__ Wk,
    const __bf16* __restrict__ Wv,
    const float* __restrict__ bq, const float* __restrict__ bk,
    const float* __restrict__ bv,
    __bf16* __restrict__ Qh, __bf16* __restrict__ Kh, __bf16* __restrict__ Vt) {
  __shared__ __align__(16) __bf16 sA[16 * DMODEL];  // 24 KB

  int mat = blockIdx.x >> 8;     // 0..2
  int mtile = blockIdx.x & 255;  // 0..255
  const __bf16* abase = xb + (size_t)mtile * 16 * DMODEL;

  // ---- stage A tile in LDS via async DMA (1536 x 16B chunks, 6/thread) ----
  {
    unsigned ldsbase = (unsigned)(unsigned long long)(void*)&sA[0];
    int tid = threadIdx.x;
#pragma unroll
    for (int i = 0; i < 6; ++i) {
      int c = tid + i * 256;
      async_copy_b128(ldsbase + c * 16, (const void*)(abase + c * 8));
    }
    asm volatile("s_wait_asynccnt 0x0" ::: "memory");
  }
  __syncthreads();

  int wave = threadIdx.x >> 5, lane = threadIdx.x & 31;
  const __bf16* W    = (mat == 0) ? Wq : (mat == 1) ? Wk : Wv;
  const float*  bias = (mat == 0) ? bq : (mat == 1) ? bk : bv;
  const __bf16* wbase = W + (size_t)(wave * 96) * DMODEL;

  v8f acc[6];
#pragma unroll
  for (int j = 0; j < 6; ++j) acc[j] = zero8();

  for (int kt = 0; kt < KT_N; ++kt) {
    int k0 = kt * 32;
    if (kt + 4 < KT_N)
      __builtin_prefetch((const void*)(wbase + (size_t)(lane & 15) * DMODEL + (kt + 4) * 32), 0, 1);
    v16bf a = load_frag(sA, DMODEL, k0, lane);  // A from LDS (ds_load_b128)
#pragma unroll
    for (int j = 0; j < 6; ++j) {
      v16bf b = load_frag(wbase + (size_t)(j * 16) * DMODEL, DMODEL, k0, lane);
      acc[j] = WMMA_BF16(a, b, acc[j]);
    }
  }

#pragma unroll
  for (int j = 0; j < 6; ++j) {
    int n = wave * 96 + j * 16 + (lane & 15);
    float bv_ = bias[n];
    int h = n >> 6, d = n & 63;
#pragma unroll
    for (int r = 0; r < 8; ++r) {
      int m   = mtile * 16 + r + ((lane >> 4) << 3);
      float v = acc[j][r] + bv_;
      int bb = m >> 11, s = m & 2047;
      int bh = bb * NHEADS + h;
      if (mat == 0)      Qh[(((size_t)bh * SEQ + s) << 6) + d] = (__bf16)v;
      else if (mat == 1) Kh[(((size_t)bh * SEQ + s) << 6) + d] = (__bf16)v;
      else               Vt[(((size_t)bh << 6) + d) * SEQ + s] = (__bf16)v;  // V transposed
    }
  }
}

// ---------------------------------------------------------------------------
// Fused attention: one workgroup = (b,h, 16-query tile).
// LDS: fp32 scores [16][2048] | bf16 probs [16][2048] | rowscale[16] | hist[8][256]
// ---------------------------------------------------------------------------
#define ATTN_LDS_BYTES (131072 + 65536 + 64 + 8192)

__global__ void attn_kernel(const __bf16* __restrict__ Qh, const __bf16* __restrict__ Kh,
                            const __bf16* __restrict__ Vt, __bf16* __restrict__ Ob) {
  extern __shared__ char smem[];
  float*    sc       = (float*)smem;                   // 16*2048 f32
  __bf16*   pe       = (__bf16*)(smem + 131072);       // 16*2048 bf16 (unnormalized exp)
  float*    rowscale = (float*)(smem + 131072 + 65536);
  unsigned* hist     = (unsigned*)(smem + 131072 + 65536 + 64);

  int wave = threadIdx.x >> 5, lane = threadIdx.x & 31;
  int bhid = blockIdx.x >> 7;          // 0..23
  int qt   = blockIdx.x & 127;         // 0..127
  int b_   = bhid / NHEADS, h_ = bhid % NHEADS;

  // ---- Phase 1: scores = (Q K^T) * 0.125 into LDS (pipelined K tiles) ------
  const __bf16* Qbase = Qh + (((size_t)bhid * SEQ) + qt * 16) * HDIM;  // 16 x 64
  const __bf16* Kbase = Kh + ((size_t)bhid * SEQ) * HDIM;              // 2048 x 64
  v16bf a0 = load_frag(Qbase, HDIM, 0, lane);
  v16bf a1 = load_frag(Qbase, HDIM, 32, lane);
  int cbase = wave * 256;

  v16bf b0c = load_frag(Kbase + (size_t)cbase * HDIM, HDIM, 0, lane);
  v16bf b1c = load_frag(Kbase + (size_t)cbase * HDIM, HDIM, 32, lane);
  for (int tt = 0; tt < 16; ++tt) {
    int col0 = cbase + tt * 16;
    if (tt + 2 < 16)
      __builtin_prefetch((const void*)(Kbase + (size_t)(col0 + 32 + (lane & 15)) * HDIM), 0, 1);
    int coln = cbase + ((tt + 1) & 15) * 16;
    v16bf b0n = load_frag(Kbase + (size_t)coln * HDIM, HDIM, 0, lane);
    v16bf b1n = load_frag(Kbase + (size_t)coln * HDIM, HDIM, 32, lane);

    v8f c = zero8();
    c = WMMA_BF16(a0, b0c, c);
    c = WMMA_BF16(a1, b1c, c);
#pragma unroll
    for (int r = 0; r < 8; ++r)
      sc[(r + ((lane >> 4) << 3)) * SEQ + col0 + (lane & 15)] = c[r] * 0.125f;

    b0c = b0n;
    b1c = b1n;
  }
  __syncthreads();

  // ---- Phase 2: per-row radix-select threshold (k-th largest) + softmax ----
  for (int rr = 0; rr < 2; ++rr) {
    int row = wave * 2 + rr;
    float*    srow = sc + (size_t)row * SEQ;
    unsigned* hst  = hist + wave * 256;
    unsigned  prefix = 0;
    int       want = KKEEP;
    for (int pass = 3; pass >= 0; --pass) {
      int sh = pass * 8;
      for (int i = lane; i < 256; i += 32) hst[i] = 0u;
      asm volatile("s_wait_dscnt 0x0" ::: "memory");
      for (int j = lane; j < SEQ; j += 32) {
        unsigned key = fkey(srow[j]);
        unsigned hs  = (pass == 3) ? 0u : (key >> (sh + 8));
        unsigned ps  = (pass == 3) ? 0u : (prefix >> (sh + 8));
        if (hs == ps) atomicAdd(&hst[(key >> sh) & 255u], 1u);
      }
      asm volatile("s_wait_dscnt 0x0" ::: "memory");
      unsigned cum = 0, chosen = 0;  // uniform scan (all lanes identical)
      for (int bn = 255; bn >= 0; --bn) {
        unsigned c = hst[bn];
        if (cum + c >= (unsigned)want) { chosen = (unsigned)bn; break; }
        cum += c;
      }
      want -= (int)cum;
      prefix |= chosen << sh;
    }
    unsigned thresh = prefix;  // key of k-th largest; keep key >= thresh

    float m = -3.0e38f;
    for (int j = lane; j < SEQ; j += 32) {
      float v = srow[j];
      if (fkey(v) >= thresh) m = fmaxf(m, v);
    }
#pragma unroll
    for (int o = 16; o; o >>= 1) m = fmaxf(m, __shfl_xor(m, o, 32));

    float ssum = 0.0f;
    __bf16* prow = pe + (size_t)row * SEQ;
    for (int j = lane; j < SEQ; j += 32) {
      float v = srow[j];
      float e = (fkey(v) >= thresh) ? __expf(v - m) : 0.0f;
      prow[j] = (__bf16)e;
      ssum += e;
    }
#pragma unroll
    for (int o = 16; o; o >>= 1) ssum += __shfl_xor(ssum, o, 32);
    if (lane == 0) rowscale[row] = 1.0f / ssum;
  }
  __syncthreads();

  // ---- Phase 3: O = P V (waves: 4 d-tiles x 2 K-halves), pipelined ---------
  int dtile = wave & 3, khalf = wave >> 2;
  v8f acc = zero8();
  const __bf16* vtb = Vt + ((size_t)bhid * HDIM + dtile * 16) * SEQ;  // rows = d, ld = SEQ
  int kstart = khalf * 1024;

  v16bf pa_c = load_frag(pe, SEQ, kstart, lane);   // probs (LDS)
  v16bf vb_c = load_frag(vtb, SEQ, kstart, lane);  // V^T (global)
  for (int t = 0; t < 32; ++t) {
    if (t + 2 < 32)
      __builtin_prefetch((const void*)(vtb + (size_t)(lane & 15) * SEQ + kstart + (t + 2) * 32), 0, 1);
    int kn = kstart + ((t + 1) & 31) * 32;
    v16bf pa_n = load_frag(pe, SEQ, kn, lane);
    v16bf vb_n = load_frag(vtb, SEQ, kn, lane);

    acc = WMMA_BF16(pa_c, vb_c, acc);

    pa_c = pa_n;
    vb_c = vb_n;
  }

  float* red = sc;  // reuse score LDS for cross-wave reduction
  if (khalf == 1) {
#pragma unroll
    for (int r = 0; r < 8; ++r) red[dtile * 256 + r * 32 + lane] = acc[r];
  }
  __syncthreads();
  if (khalf == 0) {
#pragma unroll
    for (int r = 0; r < 8; ++r) {
      float v   = acc[r] + red[dtile * 256 + r * 32 + lane];
      int  mrow = r + ((lane >> 4) << 3);
      v *= rowscale[mrow];
      int s = qt * 16 + mrow;
      int n = h_ * HDIM + dtile * 16 + (lane & 15);
      size_t mglob = (size_t)b_ * SEQ + s;
      Ob[mglob * DMODEL + n] = (__bf16)v;
    }
  }
}

// ---------------------------------------------------------------------------
// Output projection: out = O @ Wo^T + bo (fp32 result).
// One block = one mtile, async-staged LDS A tile, wave = 96-wide N strip.
// blocks: 256
// ---------------------------------------------------------------------------
__global__ void __launch_bounds__(256) oproj_kernel(
    const __bf16* __restrict__ Ob, const __bf16* __restrict__ Wo,
    const float* __restrict__ bo, float* __restrict__ out) {
  __shared__ __align__(16) __bf16 sA[16 * DMODEL];  // 24 KB

  int mtile = blockIdx.x;
  const __bf16* abase = Ob + (size_t)mtile * 16 * DMODEL;

  {
    unsigned ldsbase = (unsigned)(unsigned long long)(void*)&sA[0];
    int tid = threadIdx.x;
#pragma unroll
    for (int i = 0; i < 6; ++i) {
      int c = tid + i * 256;
      async_copy_b128(ldsbase + c * 16, (const void*)(abase + c * 8));
    }
    asm volatile("s_wait_asynccnt 0x0" ::: "memory");
  }
  __syncthreads();

  int wave = threadIdx.x >> 5, lane = threadIdx.x & 31;
  const __bf16* wbase = Wo + (size_t)(wave * 96) * DMODEL;

  v8f acc[6];
#pragma unroll
  for (int j = 0; j < 6; ++j) acc[j] = zero8();

  for (int kt = 0; kt < KT_N; ++kt) {
    int k0 = kt * 32;
    if (kt + 4 < KT_N)
      __builtin_prefetch((const void*)(wbase + (size_t)(lane & 15) * DMODEL + (kt + 4) * 32), 0, 1);
    v16bf a = load_frag(sA, DMODEL, k0, lane);
#pragma unroll
    for (int j = 0; j < 6; ++j) {
      v16bf b = load_frag(wbase + (size_t)(j * 16) * DMODEL, DMODEL, k0, lane);
      acc[j] = WMMA_BF16(a, b, acc[j]);
    }
  }

#pragma unroll
  for (int j = 0; j < 6; ++j) {
    int n = wave * 96 + j * 16 + (lane & 15);
    float bias = bo[n];
#pragma unroll
    for (int r = 0; r < 8; ++r) {
      int m = mtile * 16 + r + ((lane >> 4) << 3);
      out[(size_t)m * DMODEL + n] = acc[j][r] + bias;
    }
  }
}

// ---------------------------------------------------------------------------
// Host launcher
// ---------------------------------------------------------------------------
extern "C" void kernel_launch(void* const* d_in, const int* in_sizes, int n_in,
                              void* d_out, int out_size, void* d_ws, size_t ws_size,
                              hipStream_t stream) {
  (void)in_sizes; (void)n_in; (void)out_size; (void)ws_size;

  const float* x  = (const float*)d_in[0];
  const float* Wq = (const float*)d_in[1];
  const float* bq = (const float*)d_in[2];
  const float* Wk = (const float*)d_in[3];
  const float* bk = (const float*)d_in[4];
  const float* Wv = (const float*)d_in[5];
  const float* bv = (const float*)d_in[6];
  const float* Wo = (const float*)d_in[7];
  const float* bo = (const float*)d_in[8];
  float* out = (float*)d_out;

  // workspace layout (bytes)
  char* ws = (char*)d_ws;
  const size_t SZ_X  = (size_t)MROWS * DMODEL * 2;   // 6 MB bf16
  const size_t SZ_W  = (size_t)DMODEL * DMODEL * 2;  // 1.125 MB bf16
  __bf16* xb  = (__bf16*)(ws);
  __bf16* Wqb = (__bf16*)(ws + SZ_X);
  __bf16* Wkb = (__bf16*)(ws + SZ_X + SZ_W);
  __bf16* Wvb = (__bf16*)(ws + SZ_X + 2 * SZ_W);
  __bf16* Wob = (__bf16*)(ws + SZ_X + 3 * SZ_W);
  __bf16* Qh  = (__bf16*)(ws + SZ_X + 4 * SZ_W);
  __bf16* Kh  = (__bf16*)(ws + 2 * SZ_X + 4 * SZ_W);
  __bf16* Vt  = (__bf16*)(ws + 3 * SZ_X + 4 * SZ_W);
  __bf16* Ob  = (__bf16*)(ws + 4 * SZ_X + 4 * SZ_W);

  // 1) fp32 -> bf16 conversions
  {
    int nx = MROWS * DMODEL;             // 3,145,728
    int nw = DMODEL * DMODEL;            // 589,824
    cvt_bf16_kernel<<<(nx + 255) / 256, 256, 0, stream>>>(x, xb, nx);
    cvt_bf16_kernel<<<(nw + 255) / 256, 256, 0, stream>>>(Wq, Wqb, nw);
    cvt_bf16_kernel<<<(nw + 255) / 256, 256, 0, stream>>>(Wk, Wkb, nw);
    cvt_bf16_kernel<<<(nw + 255) / 256, 256, 0, stream>>>(Wv, Wvb, nw);
    cvt_bf16_kernel<<<(nw + 255) / 256, 256, 0, stream>>>(Wo, Wob, nw);
  }

  // 2) QKV projection (async LDS A-tile + WMMA bf16)
  qkv_kernel<<<768, 256, 0, stream>>>(xb, Wqb, Wkb, Wvb, bq, bk, bv, Qh, Kh, Vt);

  // 3) fused sparse attention (scores never leave LDS)
  (void)hipFuncSetAttribute((const void*)attn_kernel,
                            hipFuncAttributeMaxDynamicSharedMemorySize, ATTN_LDS_BYTES);
  attn_kernel<<<BATCH * NHEADS * (SEQ / 16), 256, ATTN_LDS_BYTES, stream>>>(Qh, Kh, Vt, Ob);

  // 4) output projection (async LDS A-tile + WMMA bf16) -> fp32
  oproj_kernel<<<256, 256, 0, stream>>>(Ob, Wob, bo, out);
}
